// GNBlock_7035156431516
// MI455X (gfx1250) — matrix operand order repaired and checked by reading, not compiled, not run on hardware
//
#include <hip/hip_runtime.h>
#include <hip/hip_bf16.h>

// ---------------------------------------------------------------------------
// GN block for MI455X (gfx1250): bf16 WMMA MLPs, LDS-resident activations,
// async double-buffered weight staging (global_load_async_to_lds_b128).
// 512-thread blocks: 8 row strips x 2 column groups -> 8 accumulators/wave.
// ---------------------------------------------------------------------------

#define NNODES 20000
#define NEDGES 320000
#define FDIM   128
#define HDIM   256
#define NTHR   512

typedef __attribute__((ext_vector_type(16))) __bf16 v16bf;
typedef __attribute__((ext_vector_type(8)))  float  v8f;

union FragAB {
  v16bf v;
  uint4 q[2];
  unsigned short u[16];
};

__device__ __forceinline__ unsigned short f2bf(float f) {
  unsigned int u = __float_as_uint(f);
  unsigned int r = (u + 0x7FFFu + ((u >> 16) & 1u)) >> 16;  // RNE
  return (unsigned short)r;
}

// Async copy of a contiguous region global -> LDS, 16B granules, NTHR threads.
__device__ __forceinline__ void async_copy_b128(unsigned short* ldst,
                                                const unsigned short* gsrc,
                                                int bytes) {
  unsigned int lbase = (unsigned int)(uintptr_t)ldst;  // AS3 offset = low 32 bits
  unsigned long long gbase = (unsigned long long)(uintptr_t)gsrc;
  for (int i = (int)threadIdx.x * 16; i < bytes; i += NTHR * 16) {
    asm volatile("global_load_async_to_lds_b128 %0, %1, off"
                 :: "v"(lbase + (unsigned int)i),
                    "v"(gbase + (unsigned long long)i)
                 : "memory");
  }
}
__device__ __forceinline__ void async_wait_all() {
  asm volatile("s_wait_asynccnt 0x0" ::: "memory");
}

// ---------------------------------------------------------------------------
// Weight repack: fp32 row-major W[K][N] -> bf16 WMMA-B-fragment layout.
// Lane l, tile (kt,nt) consumes W[kt*32 + (l>>4)*8 + (j&7) + (j>>3)*16][nt*16 + (l&15)],
// j = 0..15 -> two contiguous b128 loads per lane.
// ---------------------------------------------------------------------------
__global__ void pack_w_kernel(const float* __restrict__ w,
                              unsigned short* __restrict__ dst, int K, int N) {
  int tid = blockIdx.x * blockDim.x + threadIdx.x;
  int numNt = N >> 4;
  int numKt = K >> 5;
  int total = numKt * numNt * 32;
  if (tid >= total) return;
  int lane = tid & 31;
  int rem  = tid >> 5;
  int nt   = rem % numNt;
  int kt   = rem / numNt;
  int n     = nt * 16 + (lane & 15);
  int kbase = kt * 32 + (lane >> 4) * 8;
  unsigned short* d = dst + (size_t)tid * 16;
#pragma unroll
  for (int j = 0; j < 16; ++j)
    d[j] = f2bf(w[(size_t)(kbase + (j & 7) + ((j >> 3) << 4)) * N + n]);
}

// ---------------------------------------------------------------------------
// Hidden layer, M = 128. 16 waves: strip = wave&7 (16 rows each),
// colgrp = wave>>3 (half of the N-tiles each) -> 8 accumulators per wave.
// Weights streamed into LDS double buffer (2 K-tiles per chunk) via async DMA.
// ---------------------------------------------------------------------------
template <int K, int Nout, int RELU>
__device__ __forceinline__ void mlp_layer(
    const unsigned short* __restrict__ inLds, int inStride,
    unsigned short* __restrict__ outLds, int outStride,
    unsigned short* __restrict__ bbuf,               // 2 chunks
    const unsigned short* __restrict__ wp,
    const float* __restrict__ bias)
{
  constexpr int numNt   = Nout >> 4;
  constexpr int ntGrp   = numNt >> 1;                // per-wave N-tiles
  constexpr int numKt   = K >> 5;
  constexpr int sliceUS = numNt * 512;               // ushorts per K-tile
  constexpr int chunkUS = 2 * sliceUS;               // 2 K-tiles per stage
  constexpr int numCh   = numKt >> 1;

  const int lane   = threadIdx.x & 31;
  const int wave   = threadIdx.x >> 5;
  const int strip  = wave & 7;                       // row strip 0..7
  const int colgrp = wave >> 3;                      // N half 0..1
  const int m_l    = lane & 15;
  const int khalf  = lane >> 4;
  const unsigned short* arow = inLds + (size_t)(strip * 16 + m_l) * inStride;

  v8f acc[ntGrp];
#pragma unroll
  for (int u = 0; u < ntGrp; ++u)
    acc[u] = (v8f){0.f, 0.f, 0.f, 0.f, 0.f, 0.f, 0.f, 0.f};

  // Stage chunk 0.
  async_copy_b128(bbuf, wp, chunkUS * 2);
  async_wait_all();
  __syncthreads();

  for (int ch = 0; ch < numCh; ++ch) {
    const unsigned short* bcur = bbuf + (ch & 1) * chunkUS;
    if (ch + 1 < numCh)
      async_copy_b128(bbuf + ((ch + 1) & 1) * chunkUS,
                      wp + (size_t)(ch + 1) * chunkUS, chunkUS * 2);
#pragma unroll
    for (int kk = 0; kk < 2; ++kk) {
      const int kt = ch * 2 + kk;
      FragAB a;
      a.q[0] = *(const uint4*)(arow + kt * 32 + khalf * 8);
      a.q[1] = *(const uint4*)(arow + kt * 32 + 16 + khalf * 8);
      const unsigned short* bk =
          bcur + kk * sliceUS + ((size_t)(colgrp * ntGrp) * 32 + lane) * 16;
#pragma unroll
      for (int u = 0; u < ntGrp; ++u) {
        FragAB b;
        const unsigned short* bp = bk + (size_t)u * 512;
        b.q[0] = *(const uint4*)bp;
        b.q[1] = *(const uint4*)(bp + 8);
        acc[u] = __builtin_amdgcn_wmma_f32_16x16x32_bf16(
            false, a.v, false, b.v, (short)0, acc[u], false, false);
      }
    }
    async_wait_all();
    __syncthreads();
  }

#pragma unroll
  for (int u = 0; u < ntGrp; ++u) {
    int ocol = (colgrp * ntGrp + u) * 16 + m_l;
    float bn = bias[ocol];
#pragma unroll
    for (int v = 0; v < 8; ++v) {
      float x = acc[u][v] + bn;
      if (RELU) x = fmaxf(x, 0.f);
      int orow = strip * 16 + khalf * 8 + v;
      outLds[(size_t)orow * outStride + ocol] = f2bf(x);
    }
  }
}

// ---------------------------------------------------------------------------
// Edge-MLP final layer: 256 -> 128; writes e_new, atomic segment-sum into agg,
// block-local e_sum reduction in LDS.
// ---------------------------------------------------------------------------
__device__ __forceinline__ void edge_final_layer(
    const unsigned short* __restrict__ inLds, int inStride,
    unsigned short* __restrict__ bbuf,
    const unsigned short* __restrict__ wp, const float* __restrict__ bias,
    int eBase, const int* __restrict__ rcv_s,
    float* __restrict__ out_e, float* __restrict__ agg,
    float* __restrict__ esum_s)
{
  constexpr int numNt   = FDIM >> 4;   // 8
  constexpr int ntGrp   = numNt >> 1;  // 4
  constexpr int sliceUS = numNt * 512;
  constexpr int chunkUS = 2 * sliceUS;
  constexpr int numCh   = (HDIM >> 5) >> 1;   // 4

  const int lane   = threadIdx.x & 31;
  const int wave   = threadIdx.x >> 5;
  const int strip  = wave & 7;
  const int colgrp = wave >> 3;
  const int m_l    = lane & 15;
  const int khalf  = lane >> 4;
  const unsigned short* arow = inLds + (size_t)(strip * 16 + m_l) * inStride;

  v8f acc[ntGrp];
#pragma unroll
  for (int u = 0; u < ntGrp; ++u)
    acc[u] = (v8f){0.f, 0.f, 0.f, 0.f, 0.f, 0.f, 0.f, 0.f};

  async_copy_b128(bbuf, wp, chunkUS * 2);
  async_wait_all();
  __syncthreads();

  for (int ch = 0; ch < numCh; ++ch) {
    const unsigned short* bcur = bbuf + (ch & 1) * chunkUS;
    if (ch + 1 < numCh)
      async_copy_b128(bbuf + ((ch + 1) & 1) * chunkUS,
                      wp + (size_t)(ch + 1) * chunkUS, chunkUS * 2);
#pragma unroll
    for (int kk = 0; kk < 2; ++kk) {
      const int kt = ch * 2 + kk;
      FragAB a;
      a.q[0] = *(const uint4*)(arow + kt * 32 + khalf * 8);
      a.q[1] = *(const uint4*)(arow + kt * 32 + 16 + khalf * 8);
      const unsigned short* bk =
          bcur + kk * sliceUS + ((size_t)(colgrp * ntGrp) * 32 + lane) * 16;
#pragma unroll
      for (int u = 0; u < ntGrp; ++u) {
        FragAB b;
        const unsigned short* bp = bk + (size_t)u * 512;
        b.q[0] = *(const uint4*)bp;
        b.q[1] = *(const uint4*)(bp + 8);
        acc[u] = __builtin_amdgcn_wmma_f32_16x16x32_bf16(
            false, a.v, false, b.v, (short)0, acc[u], false, false);
      }
    }
    async_wait_all();
    __syncthreads();
  }

#pragma unroll
  for (int u = 0; u < ntGrp; ++u) {
    int ocol = (colgrp * ntGrp + u) * 16 + m_l;
    float bn = bias[ocol];
    float lsum = 0.f;
#pragma unroll
    for (int v = 0; v < 8; ++v) {
      int m = strip * 16 + khalf * 8 + v;
      float x = acc[u][v] + bn;
      out_e[(size_t)(eBase + m) * FDIM + ocol] = x;
      atomicAdd(&agg[(size_t)rcv_s[m] * FDIM + ocol], x);
      lsum += x;
    }
    atomicAdd(&esum_s[ocol], lsum);   // ds_add_f32
  }
}

// ---------------------------------------------------------------------------
// Node-MLP final layer: 256 -> 128; writes n_new, block-local n_sum reduction.
// ---------------------------------------------------------------------------
__device__ __forceinline__ void node_final_layer(
    const unsigned short* __restrict__ inLds, int inStride,
    unsigned short* __restrict__ bbuf,
    const unsigned short* __restrict__ wp, const float* __restrict__ bias,
    int nBase, float* __restrict__ out_n, float* __restrict__ nsum_s)
{
  constexpr int numNt   = FDIM >> 4;
  constexpr int ntGrp   = numNt >> 1;
  constexpr int sliceUS = numNt * 512;
  constexpr int chunkUS = 2 * sliceUS;
  constexpr int numCh   = (HDIM >> 5) >> 1;

  const int lane   = threadIdx.x & 31;
  const int wave   = threadIdx.x >> 5;
  const int strip  = wave & 7;
  const int colgrp = wave >> 3;
  const int m_l    = lane & 15;
  const int khalf  = lane >> 4;
  const unsigned short* arow = inLds + (size_t)(strip * 16 + m_l) * inStride;

  v8f acc[ntGrp];
#pragma unroll
  for (int u = 0; u < ntGrp; ++u)
    acc[u] = (v8f){0.f, 0.f, 0.f, 0.f, 0.f, 0.f, 0.f, 0.f};

  async_copy_b128(bbuf, wp, chunkUS * 2);
  async_wait_all();
  __syncthreads();

  for (int ch = 0; ch < numCh; ++ch) {
    const unsigned short* bcur = bbuf + (ch & 1) * chunkUS;
    if (ch + 1 < numCh)
      async_copy_b128(bbuf + ((ch + 1) & 1) * chunkUS,
                      wp + (size_t)(ch + 1) * chunkUS, chunkUS * 2);
#pragma unroll
    for (int kk = 0; kk < 2; ++kk) {
      const int kt = ch * 2 + kk;
      FragAB a;
      a.q[0] = *(const uint4*)(arow + kt * 32 + khalf * 8);
      a.q[1] = *(const uint4*)(arow + kt * 32 + 16 + khalf * 8);
      const unsigned short* bk =
          bcur + kk * sliceUS + ((size_t)(colgrp * ntGrp) * 32 + lane) * 16;
#pragma unroll
      for (int u = 0; u < ntGrp; ++u) {
        FragAB b;
        const unsigned short* bp = bk + (size_t)u * 512;
        b.q[0] = *(const uint4*)bp;
        b.q[1] = *(const uint4*)(bp + 8);
        acc[u] = __builtin_amdgcn_wmma_f32_16x16x32_bf16(
            false, a.v, false, b.v, (short)0, acc[u], false, false);
      }
    }
    async_wait_all();
    __syncthreads();
  }

#pragma unroll
  for (int u = 0; u < ntGrp; ++u) {
    int ocol = (colgrp * ntGrp + u) * 16 + m_l;
    float bn = bias[ocol];
    float lsum = 0.f;
#pragma unroll
    for (int v = 0; v < 8; ++v) {
      int m  = strip * 16 + khalf * 8 + v;
      int nd = nBase + m;
      if (nd < NNODES) {
        float x = acc[u][v] + bn;
        out_n[(size_t)nd * FDIM + ocol] = x;
        lsum += x;
      }
    }
    atomicAdd(&nsum_s[ocol], lsum);
  }
}

// ---------------------------------------------------------------------------
// Edge kernel: 128 edges/block, 512 threads, one workgroup/WGP (~271 KB LDS).
// ---------------------------------------------------------------------------
#define XS_STR 528   // 512 + 16 pad
#define HS_STR 272   // 256 + 16 pad
#define BBUF_US (2 * 2 * 16 * 512)   // 2 chunks x 2 K-tiles x 16 Nt -> 64 KB
#define EDGE_SMEM (128 * XS_STR * 2 + 128 * HS_STR * 2 + BBUF_US * 2 + 128 * 4 + 128 * 4)

__global__ __launch_bounds__(NTHR) void edge_kernel(
    const float* __restrict__ g, const float* __restrict__ nodes,
    const float* __restrict__ edges, const int* __restrict__ senders,
    const int* __restrict__ receivers,
    const unsigned short* __restrict__ w1p, const unsigned short* __restrict__ w2p,
    const unsigned short* __restrict__ w3p, const unsigned short* __restrict__ w4p,
    const float* __restrict__ b1, const float* __restrict__ b2,
    const float* __restrict__ b3, const float* __restrict__ b4,
    float* __restrict__ out_e, float* __restrict__ agg,
    float* __restrict__ esum_g)
{
  extern __shared__ unsigned char smem[];
  unsigned short* Xs     = (unsigned short*)smem;            // 128*528 bf16
  unsigned short* Hs     = Xs + 128 * XS_STR;                // 128*272 bf16
  unsigned short* Bb     = Hs + 128 * HS_STR;                // 64 KB weight pipe
  int*            rcv_s  = (int*)(Bb + BBUF_US);             // 128
  float*          esum_s = (float*)(rcv_s + 128);            // 128

  const int tid   = threadIdx.x;
  const int eBase = blockIdx.x * 128;

  // Stage concat input [g | n[snd] | n[rcv] | e] as bf16 into X.
  {
    int r   = tid >> 2;          // 0..127
    int seg = tid & 3;           // 0..3 : one 128-wide segment each
    int e   = eBase + r;
    int snd = senders[e];
    int rv  = receivers[e];
    if (seg == 0) rcv_s[r] = rv;
    const float* src;
    if (seg == 0)      src = g;
    else if (seg == 1) src = nodes + (size_t)snd * FDIM;
    else if (seg == 2) src = nodes + (size_t)rv * FDIM;
    else               src = edges + (size_t)e * FDIM;
    unsigned short* dst = Xs + (size_t)r * XS_STR + seg * FDIM;
    for (int c = 0; c < FDIM; c += 4) {
      float4 f = *(const float4*)(src + c);
      unsigned int p0 = (unsigned int)f2bf(f.x) | ((unsigned int)f2bf(f.y) << 16);
      unsigned int p1 = (unsigned int)f2bf(f.z) | ((unsigned int)f2bf(f.w) << 16);
      *(uint2*)(dst + c) = make_uint2(p0, p1);
    }
  }
  if (tid < FDIM) esum_s[tid] = 0.f;
  __syncthreads();

  mlp_layer<512, HDIM, 1>(Xs, XS_STR, Hs, HS_STR, Bb, w1p, b1);
  __syncthreads();
  mlp_layer<HDIM, HDIM, 1>(Hs, HS_STR, Xs, HS_STR, Bb, w2p, b2);
  __syncthreads();
  mlp_layer<HDIM, HDIM, 1>(Xs, HS_STR, Hs, HS_STR, Bb, w3p, b3);
  __syncthreads();
  edge_final_layer(Hs, HS_STR, Bb, w4p, b4, eBase, rcv_s, out_e, agg, esum_s);
  __syncthreads();
  if (tid < FDIM) atomicAdd(&esum_g[tid], esum_s[tid]);
}

// ---------------------------------------------------------------------------
// Node kernel: 128 nodes/block, 512 threads. Input [g | nodes | agg] (K=384).
// ---------------------------------------------------------------------------
#define XN_STR 400   // 384 + 16 pad
#define NODE_SMEM (128 * XN_STR * 2 + 128 * HS_STR * 2 + BBUF_US * 2 + 128 * 4)

__global__ __launch_bounds__(NTHR) void node_kernel(
    const float* __restrict__ g, const float* __restrict__ nodes,
    const float* __restrict__ agg,
    const unsigned short* __restrict__ w1p, const unsigned short* __restrict__ w2p,
    const unsigned short* __restrict__ w3p, const unsigned short* __restrict__ w4p,
    const float* __restrict__ b1, const float* __restrict__ b2,
    const float* __restrict__ b3, const float* __restrict__ b4,
    float* __restrict__ out_n, float* __restrict__ nsum_g)
{
  extern __shared__ unsigned char smem[];
  unsigned short* Xs     = (unsigned short*)smem;          // 128*400 bf16
  unsigned short* Hs     = Xs + 128 * XN_STR;              // 128*272 bf16
  unsigned short* Bb     = Hs + 128 * HS_STR;              // 64 KB weight pipe
  float*          nsum_s = (float*)(Bb + BBUF_US);         // 128

  const int tid   = threadIdx.x;
  const int nBase = blockIdx.x * 128;

  {
    int r   = tid >> 2;
    int seg = tid & 3;           // segs 0..2 active
    int nd  = nBase + r;
    bool valid = (nd < NNODES);
    if (seg < 3) {
      const float* src = (seg == 0) ? g
                       : (seg == 1) ? nodes + (size_t)nd * FDIM
                                    : agg + (size_t)nd * FDIM;
      unsigned short* dst = Xs + (size_t)r * XN_STR + seg * FDIM;
      for (int c = 0; c < FDIM; c += 4) {
        float4 f = valid ? *(const float4*)(src + c)
                         : make_float4(0.f, 0.f, 0.f, 0.f);
        unsigned int p0 = (unsigned int)f2bf(f.x) | ((unsigned int)f2bf(f.y) << 16);
        unsigned int p1 = (unsigned int)f2bf(f.z) | ((unsigned int)f2bf(f.w) << 16);
        *(uint2*)(dst + c) = make_uint2(p0, p1);
      }
    }
  }
  if (tid < FDIM) nsum_s[tid] = 0.f;
  __syncthreads();

  mlp_layer<384, HDIM, 1>(Xs, XN_STR, Hs, HS_STR, Bb, w1p, b1);
  __syncthreads();
  mlp_layer<HDIM, HDIM, 1>(Hs, HS_STR, Xs, HS_STR, Bb, w2p, b2);
  __syncthreads();
  mlp_layer<HDIM, HDIM, 1>(Xs, HS_STR, Hs, HS_STR, Bb, w3p, b3);
  __syncthreads();
  node_final_layer(Hs, HS_STR, Bb, w4p, b4, nBase, out_n, nsum_s);
  __syncthreads();
  if (tid < FDIM) atomicAdd(&nsum_g[tid], nsum_s[tid]);
}

// ---------------------------------------------------------------------------
// Graph MLP: single row, single block, plain fp32 (WMMA pointless at M=1).
// ---------------------------------------------------------------------------
__global__ __launch_bounds__(256) void graph_kernel(
    const float* __restrict__ g, const float* __restrict__ nsum,
    const float* __restrict__ esum,
    const float* __restrict__ w1, const float* __restrict__ bb1,
    const float* __restrict__ w2, const float* __restrict__ bb2,
    const float* __restrict__ w3, const float* __restrict__ bb3,
    const float* __restrict__ w4, const float* __restrict__ bb4,
    float* __restrict__ out_g)
{
  __shared__ float x[384];
  __shared__ float ha[256];
  __shared__ float hb[256];
  int t = threadIdx.x;
  if (t < 128) {
    x[t]       = g[t];
    x[128 + t] = nsum[t];
    x[256 + t] = esum[t];
  }
  __syncthreads();
  float a = bb1[t];
  for (int k = 0; k < 384; ++k) a += x[k] * w1[k * 256 + t];
  ha[t] = fmaxf(a, 0.f);
  __syncthreads();
  a = bb2[t];
  for (int k = 0; k < 256; ++k) a += ha[k] * w2[k * 256 + t];
  hb[t] = fmaxf(a, 0.f);
  __syncthreads();
  a = bb3[t];
  for (int k = 0; k < 256; ++k) a += hb[k] * w3[k * 256 + t];
  ha[t] = fmaxf(a, 0.f);
  __syncthreads();
  if (t < 128) {
    a = bb4[t];
    for (int k = 0; k < 256; ++k) a += ha[k] * w4[k * 128 + t];
    out_g[t] = a;
  }
}

// ---------------------------------------------------------------------------
// Launch
// ---------------------------------------------------------------------------
extern "C" void kernel_launch(void* const* d_in, const int* in_sizes, int n_in,
                              void* d_out, int out_size, void* d_ws, size_t ws_size,
                              hipStream_t stream) {
  (void)in_sizes; (void)n_in; (void)out_size; (void)ws_size;

  const float* g         = (const float*)d_in[0];
  const float* nodes     = (const float*)d_in[1];
  const float* edges     = (const float*)d_in[2];
  const int*   senders   = (const int*)d_in[3];
  const int*   receivers = (const int*)d_in[4];
  const float* ew[4] = {(const float*)d_in[5],  (const float*)d_in[7],
                        (const float*)d_in[9],  (const float*)d_in[11]};
  const float* eb[4] = {(const float*)d_in[6],  (const float*)d_in[8],
                        (const float*)d_in[10], (const float*)d_in[12]};
  const float* nw[4] = {(const float*)d_in[13], (const float*)d_in[15],
                        (const float*)d_in[17], (const float*)d_in[19]};
  const float* nb[4] = {(const float*)d_in[14], (const float*)d_in[16],
                        (const float*)d_in[18], (const float*)d_in[20]};
  const float* gw[4] = {(const float*)d_in[21], (const float*)d_in[23],
                        (const float*)d_in[25], (const float*)d_in[27]};
  const float* gb[4] = {(const float*)d_in[22], (const float*)d_in[24],
                        (const float*)d_in[26], (const float*)d_in[28]};

  // Workspace layout
  unsigned char* ws = (unsigned char*)d_ws;
  float* agg  = (float*)ws;  ws += (size_t)NNODES * FDIM * 4;   // 10.24 MB
  float* esum = (float*)ws;  ws += 512;
  float* nsum = (float*)ws;  ws += 512;
  unsigned short* ew1p = (unsigned short*)ws;  ws += (size_t)512 * 256 * 2;
  unsigned short* ew2p = (unsigned short*)ws;  ws += (size_t)256 * 256 * 2;
  unsigned short* ew3p = (unsigned short*)ws;  ws += (size_t)256 * 256 * 2;
  unsigned short* ew4p = (unsigned short*)ws;  ws += (size_t)256 * 128 * 2;
  unsigned short* nw1p = (unsigned short*)ws;  ws += (size_t)384 * 256 * 2;
  unsigned short* nw2p = (unsigned short*)ws;  ws += (size_t)256 * 256 * 2;
  unsigned short* nw3p = (unsigned short*)ws;  ws += (size_t)256 * 256 * 2;
  unsigned short* nw4p = (unsigned short*)ws;  ws += (size_t)256 * 128 * 2;

  float* out_e = (float*)d_out;
  float* out_n = out_e + (size_t)NEDGES * FDIM;
  float* out_g = out_n + (size_t)NNODES * FDIM;

  // Zero agg + esum + nsum (contiguous).
  hipMemsetAsync(agg, 0, (size_t)NNODES * FDIM * 4 + 1024, stream);

  // Repack weights to WMMA-B bf16 layout.
  auto pack = [&](const float* w, unsigned short* dst, int K, int N) {
    int total = (K / 32) * (N / 16) * 32;
    pack_w_kernel<<<(total + 255) / 256, 256, 0, stream>>>(w, dst, K, N);
  };
  pack(ew[0], ew1p, 512, 256);
  pack(ew[1], ew2p, 256, 256);
  pack(ew[2], ew3p, 256, 256);
  pack(ew[3], ew4p, 256, 128);
  pack(nw[0], nw1p, 384, 256);
  pack(nw[1], nw2p, 256, 256);
  pack(nw[2], nw3p, 256, 256);
  pack(nw[3], nw4p, 256, 128);

  (void)hipFuncSetAttribute((const void*)edge_kernel,
                            hipFuncAttributeMaxDynamicSharedMemorySize, EDGE_SMEM);
  (void)hipFuncSetAttribute((const void*)node_kernel,
                            hipFuncAttributeMaxDynamicSharedMemorySize, NODE_SMEM);

  edge_kernel<<<NEDGES / 128, NTHR, EDGE_SMEM, stream>>>(
      g, nodes, edges, senders, receivers,
      ew1p, ew2p, ew3p, ew4p, eb[0], eb[1], eb[2], eb[3],
      out_e, agg, esum);

  node_kernel<<<(NNODES + 127) / 128, NTHR, NODE_SMEM, stream>>>(
      g, nodes, agg,
      nw1p, nw2p, nw3p, nw4p, nb[0], nb[1], nb[2], nb[3],
      out_n, nsum);

  graph_kernel<<<1, 256, 0, stream>>>(
      g, nsum, esum,
      gw[0], gb[0], gw[1], gb[1], gw[2], gb[2], gw[3], gb[3],
      out_g);
}
